// BottleneckSparseAttention2D_12790412608009
// MI455X (gfx1250) — matrix operand order repaired and checked by reading, not compiled, not run on hardware
//
#include <hip/hip_runtime.h>
#include <hip/hip_bf16.h>

typedef __attribute__((ext_vector_type(16))) _Float16 v16h;
typedef __attribute__((ext_vector_type(8)))  _Float16 v8h;
typedef __attribute__((ext_vector_type(8)))  float    v8f;

#define WMMA(a, b, c) \
  __builtin_amdgcn_wmma_f32_16x16x32_f16(false, (a), false, (b), (short)0, (c), false, false)

// ---------------------------------------------------------------------------
// A-operand tile load (16x32 f16). ISA layout (CDNA5 ISA, 16-bit A 16x32):
//   lane<16 : row = lane,    vec[0..7] = K 0..7,   vec[8..15] = K 16..23
//   lane>=16: row = lane-16, vec[0..7] = K 8..15,  vec[8..15] = K 24..31
// Source is row-major f16 with given stride -> two aligned 16B loads.
// ---------------------------------------------------------------------------
__device__ __forceinline__ v16h load_tileA(const _Float16* base, int rowBase,
                                           int stride, int kBase) {
  const int lane = threadIdx.x & 31;
  const _Float16* p =
      base + (rowBase + (lane & 15)) * stride + kBase + ((lane >> 4) << 3);
  v8h lo = *reinterpret_cast<const v8h*>(p);
  v8h hi = *reinterpret_cast<const v8h*>(p + 16);
  return __builtin_shufflevector(lo, hi, 0, 1, 2, 3, 4, 5, 6, 7, 8, 9, 10, 11,
                                 12, 13, 14, 15);
}

// ---------------------------------------------------------------------------
// B-operand tile load (32x16 f16):
//   lane<16 : col = lane,    vec[0..7] = K 0..7,   vec[8..15] = K 16..23
//   lane>=16: col = lane-16, vec[0..7] = K 8..15,  vec[8..15] = K 24..31
// Source is W^T row-major (one output-column per row), contiguous in K.
// ---------------------------------------------------------------------------
__device__ __forceinline__ v16h load_tileB(const _Float16* base, int colBase,
                                           int stride, int kBase) {
  const int lane = threadIdx.x & 31;
  const _Float16* p =
      base + (colBase + (lane & 15)) * stride + kBase + ((lane >> 4) << 3);
  v8h lo = *reinterpret_cast<const v8h*>(p);
  v8h hi = *reinterpret_cast<const v8h*>(p + 16);
  return __builtin_shufflevector(lo, hi, 0, 1, 2, 3, 4, 5, 6, 7, 8, 9, 10, 11,
                                 12, 13, 14, 15);
}

// C-layout (16x16 f32, 8 VGPRs): VGPR r -> lanes 0-15: M=r, N=lane;
// lanes 16-31: M=r+8, N=lane-16.  Store converted to f16 LDS.
__device__ __forceinline__ void store_tile_f16(_Float16* dst, int mBase,
                                               int nBase, int stride, v8f c) {
  const int lane = threadIdx.x & 31;
  const int col  = nBase + (lane & 15);
  const int row0 = mBase + ((lane >> 4) << 3);
#pragma unroll
  for (int r = 0; r < 8; ++r) dst[(row0 + r) * stride + col] = (_Float16)c[r];
}

// ---------------------------------------------------------------------------
// Weight prep: fp32 (in,out) row-major  ->  f16 transposed (out,in) row-major
// ---------------------------------------------------------------------------
__global__ void transpose_cast_kernel(const float* __restrict__ w,
                                      _Float16* __restrict__ o, int in_dim,
                                      int out_dim) {
  int idx = blockIdx.x * blockDim.x + threadIdx.x;
  if (idx >= in_dim * out_dim) return;
  int oc = idx / in_dim;
  int ic = idx - oc * in_dim;
  o[idx] = (_Float16)w[ic * out_dim + oc];
}

// ---------------------------------------------------------------------------
// Fused per-patch transformer block.  1 workgroup = 1 patch of 64 rows.
// 256 threads = 8 wave32; wave w owns head w in the attention phase.
//
// LDS map (bytes) -- exactly 160 KB so 2 workgroups fit per 320 KB WGP:
//   [0      , 32768 )  xbuf   f32 64x128 residual stream; reused as f32 64x64
//                      output staging in phase 9
//   [32768  , 49152 )  nobuf  f16 64x128: LN output (ph 2-3, 6-7) or f16 x2 (ph 9) overlaid
//                      with attention output obuf (ph 4-5)
//   [49152  , 114688)  qkbuf  f16 8 heads x [q 64x32 | k 64x32] (hd padded to
//                      32, q pre-scaled); per-head att scores f16 64x64
//                      overlay q|k of the same head inside phase 4
//   [114688 , 131072)  vtbuf  f16 8x16x64 (V transposed per head)
//   [131072 , 163840)  scratch: featsA f16 64x64 (ph 0-1) | hbuf f16 64x256
//                      (ph 7-8)
// ---------------------------------------------------------------------------
#define SMEM_BYTES 163840

__global__ void __launch_bounds__(256)
fused_block_kernel(const float* __restrict__ feats,
                   const _Float16* __restrict__ w_preT,
                   const float* __restrict__ b_pre,
                   const float* __restrict__ g1, const float* __restrict__ b1,
                   const _Float16* __restrict__ w_qkvT,
                   const float* __restrict__ b_qkv,
                   const _Float16* __restrict__ w_projT,
                   const float* __restrict__ b_proj,
                   const float* __restrict__ g2, const float* __restrict__ b2,
                   const _Float16* __restrict__ w_m1T,
                   const float* __restrict__ b_m1,
                   const _Float16* __restrict__ w_m2T,
                   const float* __restrict__ b_m2,
                   const _Float16* __restrict__ w_postT,
                   const float* __restrict__ b_post,
                   float* __restrict__ out) {
  __shared__ __align__(128) char smem[SMEM_BYTES];
  float*    xbuf    = reinterpret_cast<float*>(smem);
  _Float16* nobuf   = reinterpret_cast<_Float16*>(smem + 32768);
  _Float16* qkbuf   = reinterpret_cast<_Float16*>(smem + 49152);
  _Float16* vtbuf   = reinterpret_cast<_Float16*>(smem + 114688);
  _Float16* scratch = reinterpret_cast<_Float16*>(smem + 131072);

  const int tid   = threadIdx.x;
  const int lane  = tid & 31;
  const int wave  = tid >> 5;
  const int patch = blockIdx.x;

  // ---- Phase 0: stage feats tile as f16 (b128 loads); zero q/k padding ----
  {
    _Float16* featsA = scratch;  // 64x64
    const float4* fsrc4 =
        reinterpret_cast<const float4*>(feats + (size_t)patch * 64 * 64);
    for (int v = tid; v < 64 * 64 / 4; v += 256) {
      float4 f = fsrc4[v];
      featsA[v * 4 + 0] = (_Float16)f.x;
      featsA[v * 4 + 1] = (_Float16)f.y;
      featsA[v * 4 + 2] = (_Float16)f.z;
      featsA[v * 4 + 3] = (_Float16)f.w;
    }
    // zero d=16..31 of every q row and k row (8 heads x 64 rows x 16)
    for (int idx = tid; idx < 8 * 64 * 16; idx += 256) {
      int h = idx >> 10, rem = idx & 1023, row = rem >> 4, j = rem & 15;
      qkbuf[h * 4096 + row * 32 + 16 + j]        = (_Float16)0.f;  // q pad
      qkbuf[h * 4096 + 2048 + row * 32 + 16 + j] = (_Float16)0.f;  // k pad
    }
  }
  __syncthreads();

  // ---- Phase 1: x2 = feats @ w_pre + b_pre  (64x128, K=64) ----------------
  {
    _Float16* featsA = scratch;
#pragma unroll
    for (int i = 0; i < 4; ++i) {
      int t = wave * 4 + i, mt = t >> 3, nt = t & 7;
      v8f c = {};
#pragma unroll
      for (int ks = 0; ks < 2; ++ks) {
        v16h a = load_tileA(featsA, mt * 16, 64, ks * 32);
        v16h b = load_tileB(w_preT, nt * 16, 64, ks * 32);
        c = WMMA(a, b, c);
      }
      int col = nt * 16 + (lane & 15);
      int row0 = mt * 16 + ((lane >> 4) << 3);
      float bias = b_pre[col];
#pragma unroll
      for (int r = 0; r < 8; ++r) xbuf[(row0 + r) * 128 + col] = c[r] + bias;
    }
  }
  __syncthreads();

  // ---- Phase 2: n1 = LN(x2)*g1 + b1 ---------------------------------------
  if (tid < 64) {
    int row = tid;
    float s = 0.f, s2 = 0.f;
    for (int cix = 0; cix < 128; ++cix) {
      float v = xbuf[row * 128 + cix];
      s += v;
      s2 += v * v;
    }
    float mu   = s * (1.f / 128.f);
    float var  = s2 * (1.f / 128.f) - mu * mu;
    float rstd = rsqrtf(var + 1e-5f);
    for (int cix = 0; cix < 128; ++cix) {
      float v = (xbuf[row * 128 + cix] - mu) * rstd;
      nobuf[row * 128 + cix] = (_Float16)(v * g1[cix] + b1[cix]);
    }
  }
  __syncthreads();

  // ---- Phase 3: qkv = n1 @ w_qkv + b_qkv (64x384, K=128); scatter ---------
  {
#pragma unroll
    for (int i = 0; i < 12; ++i) {
      int t = wave * 12 + i, mt = t / 24, nt = t % 24;
      v8f c = {};
#pragma unroll
      for (int ks = 0; ks < 4; ++ks) {
        v16h a = load_tileA(nobuf, mt * 16, 128, ks * 32);
        v16h b = load_tileB(w_qkvT, nt * 16, 128, ks * 32);
        c = WMMA(a, b, c);
      }
      int col  = nt * 16 + (lane & 15);
      int row0 = mt * 16 + ((lane >> 4) << 3);
      float bias = b_qkv[col];
      int t3 = col >> 7, rem = col & 127, h = rem >> 4, d = rem & 15;
#pragma unroll
      for (int r = 0; r < 8; ++r) {
        float v = c[r] + bias;
        int row = row0 + r;
        if (t3 == 0)
          qkbuf[h * 4096 + row * 32 + d] = (_Float16)(v * 0.25f);  // q, scaled
        else if (t3 == 1)
          qkbuf[h * 4096 + 2048 + row * 32 + d] = (_Float16)v;     // k
        else
          vtbuf[(h * 16 + d) * 64 + row] = (_Float16)v;            // v^T
      }
    }
  }
  __syncthreads();

  // ---- Phase 4: attention, wave h = head h --------------------------------
  {
    const int h = wave;
    const _Float16* qh = qkbuf + h * 4096;         // 64x32
    const _Float16* kh = qkbuf + h * 4096 + 2048;  // 64x32
    // Scores overlay this head's q|k (8 KB): safe because q/k are fully
    // consumed into VGPRs below before any score store, LDS ops are in-order
    // within a wave, and no other wave touches head h's region.
    _Float16* atth = qkbuf + h * 4096;             // f16 64x64

    v16h qa[4], kb[4];
#pragma unroll
    for (int mt = 0; mt < 4; ++mt) qa[mt] = load_tileA(qh, mt * 16, 32, 0);
#pragma unroll
    for (int nt = 0; nt < 4; ++nt) kb[nt] = load_tileB(kh, nt * 16, 32, 0);

    v8f S[4][4];
#pragma unroll
    for (int mt = 0; mt < 4; ++mt)
#pragma unroll
      for (int nt = 0; nt < 4; ++nt) {
        v8f z = {};
        S[mt][nt] = WMMA(qa[mt], kb[nt], z);
      }

    // In-register softmax over each row (row = VGPR index within lane-half;
    // columns live across 16 lanes x 4 N-tiles -> 16-lane xor reductions).
#pragma unroll
    for (int mt = 0; mt < 4; ++mt) {
#pragma unroll
      for (int r = 0; r < 8; ++r) {
        float mx = S[mt][0][r];
#pragma unroll
        for (int nt = 1; nt < 4; ++nt) mx = fmaxf(mx, S[mt][nt][r]);
#pragma unroll
        for (int off = 1; off < 16; off <<= 1)
          mx = fmaxf(mx, __shfl_xor(mx, off, 32));
        float sum = 0.f;
#pragma unroll
        for (int nt = 0; nt < 4; ++nt) {
          float e = __expf(S[mt][nt][r] - mx);
          S[mt][nt][r] = e;
          sum += e;
        }
#pragma unroll
        for (int off = 1; off < 16; off <<= 1) sum += __shfl_xor(sum, off, 32);
        float inv = 1.f / sum;
#pragma unroll
        for (int nt = 0; nt < 4; ++nt) S[mt][nt][r] *= inv;
      }
#pragma unroll
      for (int nt = 0; nt < 4; ++nt)
        store_tile_f16(atth, mt * 16, nt * 16, 64, S[mt][nt]);
    }

    // O_h = att @ V_h  (64x16, K=64); O goes to nobuf (n1 is dead now)
    const _Float16* vth = vtbuf + h * 16 * 64;
#pragma unroll
    for (int mt = 0; mt < 4; ++mt) {
      v8f c = {};
#pragma unroll
      for (int ks = 0; ks < 2; ++ks) {
        v16h a = load_tileA(atth, mt * 16, 64, ks * 32);
        v16h b = load_tileB(vth, 0, 64, ks * 32);
        c = WMMA(a, b, c);
      }
      store_tile_f16(nobuf, mt * 16, h * 16, 128, c);
    }
  }
  __syncthreads();

  // ---- Phase 5: x2 += O @ w_proj + b_proj  (64x128, K=128) ----------------
  {
#pragma unroll
    for (int i = 0; i < 4; ++i) {
      int t = wave * 4 + i, mt = t >> 3, nt = t & 7;
      v8f c = {};
#pragma unroll
      for (int ks = 0; ks < 4; ++ks) {
        v16h a = load_tileA(nobuf, mt * 16, 128, ks * 32);
        v16h b = load_tileB(w_projT, nt * 16, 128, ks * 32);
        c = WMMA(a, b, c);
      }
      int col = nt * 16 + (lane & 15);
      int row0 = mt * 16 + ((lane >> 4) << 3);
      float bias = b_proj[col];
#pragma unroll
      for (int r = 0; r < 8; ++r) xbuf[(row0 + r) * 128 + col] += c[r] + bias;
    }
  }
  __syncthreads();

  // ---- Phase 6: n2 = LN(x2)*g2 + b2 ---------------------------------------
  if (tid < 64) {
    int row = tid;
    float s = 0.f, s2 = 0.f;
    for (int cix = 0; cix < 128; ++cix) {
      float v = xbuf[row * 128 + cix];
      s += v;
      s2 += v * v;
    }
    float mu   = s * (1.f / 128.f);
    float var  = s2 * (1.f / 128.f) - mu * mu;
    float rstd = rsqrtf(var + 1e-5f);
    for (int cix = 0; cix < 128; ++cix) {
      float v = (xbuf[row * 128 + cix] - mu) * rstd;
      nobuf[row * 128 + cix] = (_Float16)(v * g2[cix] + b2[cix]);
    }
  }
  __syncthreads();

  // ---- Phase 7: hbuf = gelu(n2 @ w_m1 + b_m1)  (64x256, K=128) ------------
  {
    _Float16* hbuf = scratch;  // 64x256
#pragma unroll
    for (int i = 0; i < 8; ++i) {
      int t = wave * 8 + i, mt = t >> 4, nt = t & 15;
      v8f c = {};
#pragma unroll
      for (int ks = 0; ks < 4; ++ks) {
        v16h a = load_tileA(nobuf, mt * 16, 128, ks * 32);
        v16h b = load_tileB(w_m1T, nt * 16, 128, ks * 32);
        c = WMMA(a, b, c);
      }
      int col = nt * 16 + (lane & 15);
      int row0 = mt * 16 + ((lane >> 4) << 3);
      float bias = b_m1[col];
#pragma unroll
      for (int r = 0; r < 8; ++r) {
        float x = c[r] + bias;
        float g = 0.5f * x * (1.f + erff(x * 0.70710678118654752f));
        hbuf[(row0 + r) * 256 + col] = (_Float16)g;
      }
    }
  }
  __syncthreads();

  // ---- Phase 8: x2 += hbuf @ w_m2 + b_m2  (64x128, K=256) -----------------
  {
    _Float16* hbuf = scratch;
#pragma unroll
    for (int i = 0; i < 4; ++i) {
      int t = wave * 4 + i, mt = t >> 3, nt = t & 7;
      v8f c = {};
#pragma unroll
      for (int ks = 0; ks < 8; ++ks) {
        v16h a = load_tileA(hbuf, mt * 16, 256, ks * 32);
        v16h b = load_tileB(w_m2T, nt * 16, 256, ks * 32);
        c = WMMA(a, b, c);
      }
      int col = nt * 16 + (lane & 15);
      int row0 = mt * 16 + ((lane >> 4) << 3);
      float bias = b_m2[col];
#pragma unroll
      for (int r = 0; r < 8; ++r) xbuf[(row0 + r) * 128 + col] += c[r] + bias;
    }
  }
  __syncthreads();

  // ---- Phase 9: out = x2 @ w_post + b_post  (64x64, K=128) ----------------
  // 9a: f16 copy of x2 into nobuf
  for (int idx = tid; idx < 64 * 128; idx += 256)
    nobuf[idx] = (_Float16)xbuf[idx];
  __syncthreads();  // xbuf is now dead -> reuse as f32 output staging
  {
#pragma unroll
    for (int i = 0; i < 2; ++i) {
      int t = wave * 2 + i, mt = t >> 2, nt = t & 3;
      v8f c = {};
#pragma unroll
      for (int ks = 0; ks < 4; ++ks) {
        v16h a = load_tileA(nobuf, mt * 16, 128, ks * 32);
        v16h b = load_tileB(w_postT, nt * 16, 128, ks * 32);
        c = WMMA(a, b, c);
      }
      int col = nt * 16 + (lane & 15);
      int row0 = mt * 16 + ((lane >> 4) << 3);
      float bias = b_post[col];
#pragma unroll
      for (int r = 0; r < 8; ++r) xbuf[(row0 + r) * 64 + col] = c[r] + bias;
    }
  }
  __syncthreads();
  // 9b: fully coalesced b128 stores to HBM
  {
    float4* odst4 = reinterpret_cast<float4*>(out + (size_t)patch * 64 * 64);
    const float4* src4 = reinterpret_cast<const float4*>(xbuf);
    for (int v = tid; v < 64 * 64 / 4; v += 256) odst4[v] = src4[v];
  }
}

// ---------------------------------------------------------------------------
extern "C" void kernel_launch(void* const* d_in, const int* in_sizes, int n_in,
                              void* d_out, int out_size, void* d_ws,
                              size_t ws_size, hipStream_t stream) {
  const float* feats  = (const float*)d_in[0];
  const float* w_pre  = (const float*)d_in[1];
  const float* b_pre  = (const float*)d_in[2];
  const float* g1     = (const float*)d_in[3];
  const float* b1     = (const float*)d_in[4];
  const float* w_qkv  = (const float*)d_in[5];
  const float* b_qkv  = (const float*)d_in[6];
  const float* w_proj = (const float*)d_in[7];
  const float* b_proj = (const float*)d_in[8];
  const float* g2     = (const float*)d_in[9];
  const float* b2     = (const float*)d_in[10];
  const float* w_m1   = (const float*)d_in[11];
  const float* b_m1   = (const float*)d_in[12];
  const float* w_m2   = (const float*)d_in[13];
  const float* b_m2   = (const float*)d_in[14];
  const float* w_post = (const float*)d_in[15];
  const float* b_post = (const float*)d_in[16];
  float* out = (float*)d_out;

  // Workspace: transposed f16 weights (294,912 bytes total), L2-resident.
  _Float16* ws      = (_Float16*)d_ws;
  _Float16* w_preT  = ws;           // 128x64
  _Float16* w_qkvT  = ws + 8192;    // 384x128
  _Float16* w_projT = ws + 57344;   // 128x128
  _Float16* w_m1T   = ws + 73728;   // 256x128
  _Float16* w_m2T   = ws + 106496;  // 128x256
  _Float16* w_postT = ws + 139264;  // 64x128

  auto prep = [&](const float* w, _Float16* o, int in_dim, int out_dim) {
    int n = in_dim * out_dim;
    hipLaunchKernelGGL(transpose_cast_kernel, dim3((n + 255) / 256), dim3(256),
                       0, stream, w, o, in_dim, out_dim);
  };
  prep(w_pre, w_preT, 64, 128);
  prep(w_qkv, w_qkvT, 128, 384);
  prep(w_proj, w_projT, 128, 128);
  prep(w_m1, w_m1T, 128, 256);
  prep(w_m2, w_m2T, 256, 128);
  prep(w_post, w_postT, 128, 64);

  hipLaunchKernelGGL(fused_block_kernel, dim3(8192), dim3(256), 0, stream,
                     feats, w_preT, b_pre, g1, b1, w_qkvT, b_qkv, w_projT,
                     b_proj, g2, b2, w_m1T, b_m1, w_m2T, b_m2, w_postT, b_post,
                     out);
}
